// HalfKP_NNUE_67860483276871
// MI455X (gfx1250) — matrix operand order repaired and checked by reading, not compiled, not run on hardware
//
#include <hip/hip_runtime.h>
#include <hip/hip_bf16.h>

// ---------------------------------------------------------------------------
// Fused HalfKP-NNUE forward for MI455X (gfx1250, wave32, WMMA).
//  - Phase A: embedding-bag gather-sum from L2-resident tables (f32, b128 loads)
//  - Phase B: fc2 via v_wmma_f32_16x16x32_bf16 (K=512), fc3 via one WMMA (K=32)
//  - fc4: per-lane dot product
// Weights are converted to bf16 once per launch into d_ws by a prep kernel.
// ---------------------------------------------------------------------------

typedef __attribute__((ext_vector_type(4)))  __bf16 v4bf;
typedef __attribute__((ext_vector_type(8)))  __bf16 v8bf;
typedef __attribute__((ext_vector_type(16))) __bf16 v16bf;
typedef __attribute__((ext_vector_type(8)))  float  v8f;

#define TABLE    640
#define HIDDEN   256
#define TWO_H    512
#define MLPH     32
#define NACT     30
#define BT       16          // batch rows per block
#define THREADS  512         // 16 waves (wave32)
#define HSTRIDE  520         // bf16 elems per h row (pad: 1040B % 256B = 16 -> 4-bank skew)
#define H2STRIDE 40          // bf16 elems per h2 row (80B, 16B aligned)
#define H3STRIDE 33          // f32 elems per h3 row

__device__ __forceinline__ void add4(float4& a, const float4 b) {
    a.x += b.x; a.y += b.y; a.z += b.z; a.w += b.w;
}

__device__ __forceinline__ v4bf relu_pack4(const float4 v) {
    v4bf r;
    r[0] = (__bf16)fmaxf(v.x, 0.0f);
    r[1] = (__bf16)fmaxf(v.y, 0.0f);
    r[2] = (__bf16)fmaxf(v.z, 0.0f);
    r[3] = (__bf16)fmaxf(v.w, 0.0f);
    return r;
}

// One-shot weight conversion f32 -> bf16 into workspace.
__global__ __launch_bounds__(256) void nnue_cvt_weights(
    const float* __restrict__ fc2w, const float* __restrict__ fc3w,
    unsigned short* __restrict__ ws2, unsigned short* __restrict__ ws3)
{
    int i = blockIdx.x * blockDim.x + threadIdx.x;
    if (i < MLPH * TWO_H) {
        __bf16 v = (__bf16)fc2w[i];
        ws2[i] = *reinterpret_cast<unsigned short*>(&v);
    }
    if (i < MLPH * MLPH) {
        __bf16 v = (__bf16)fc3w[i];
        ws3[i] = *reinterpret_cast<unsigned short*>(&v);
    }
}

__global__ __launch_bounds__(THREADS) void nnue_fused(
    const int*   __restrict__ idx0,
    const int*   __restrict__ idx1,
    const float* __restrict__ w1,
    const unsigned short* __restrict__ fc2w_u,  // bf16 bits [32][512]
    const float* __restrict__ fc2b,
    const unsigned short* __restrict__ fc3w_u,  // bf16 bits [32][32]
    const float* __restrict__ fc3b,
    const float* __restrict__ fc4w,
    const float* __restrict__ fc4b,
    float* __restrict__ out)
{
    __shared__ int sidx0[BT * NACT];
    __shared__ int sidx1[BT * NACT];
    __shared__ alignas(16) __bf16 hbuf [BT * HSTRIDE];   // relu(h) tile, bf16
    __shared__ alignas(16) __bf16 h2buf[BT * H2STRIDE];  // relu(fc2) tile, bf16
    __shared__ alignas(16) float  h3buf[BT * H3STRIDE];  // relu(fc3) tile, f32

    const int tid   = threadIdx.x;
    const int wave  = tid >> 5;
    const int lane  = tid & 31;
    const int rbase = blockIdx.x * BT;

    // Stage block's indices into LDS (coalesced, then scalar-broadcast reads).
    if (tid < BT * NACT) {
        sidx0[tid] = idx0[rbase * NACT + tid];
        sidx1[tid] = idx1[rbase * NACT + tid];
    }
    __syncthreads();

    // ---------------- Phase A: embedding-bag (one wave per batch row) -------
    {
        const int row = wave;                 // 0..15
        const float* t0 = w1;                 // table 0 [640][256]
        const float* t1 = w1 + TABLE * HIDDEN;// table 1
        const int c0 = 4 * lane;              // lane owns dims [c0..c0+3], [128+c0..]

        float4 a0 = {0.f,0.f,0.f,0.f}, a1 = {0.f,0.f,0.f,0.f};
        float4 b0 = {0.f,0.f,0.f,0.f}, b1 = {0.f,0.f,0.f,0.f};

        #pragma unroll 2
        for (int j = 0; j < NACT; ++j) {
            const int i0 = sidx0[row * NACT + j];
            const int i1 = sidx1[row * NACT + j];
            const float* r0 = t0 + i0 * HIDDEN;
            const float* r1 = t1 + i1 * HIDDEN;
            float4 x0 = *(const float4*)(r0 + c0);
            float4 x1 = *(const float4*)(r0 + 128 + c0);
            float4 y0 = *(const float4*)(r1 + c0);
            float4 y1 = *(const float4*)(r1 + 128 + c0);
            add4(a0, x0); add4(a1, x1);
            add4(b0, y0); add4(b1, y1);
        }
        // relu + bf16 pack into LDS: h[row] = [relu(sum0)(256), relu(sum1)(256)]
        __bf16* hr = &hbuf[row * HSTRIDE];
        *(v4bf*)(hr + c0)       = relu_pack4(a0);
        *(v4bf*)(hr + 128 + c0) = relu_pack4(a1);
        *(v4bf*)(hr + 256 + c0) = relu_pack4(b0);
        *(v4bf*)(hr + 384 + c0) = relu_pack4(b1);
    }
    __syncthreads();

    // ---------------- Phase B: fc2 = relu(h @ fc2_w.T + b), WMMA bf16 ------
    // Waves 0/1 each own one 16-wide N tile. K = 512 -> 16 WMMA steps.
    if (wave < 2) {
        const __bf16* fc2w_bf = (const __bf16*)fc2w_u;
        const int  nb   = wave * 16;
        const bool hiL  = (lane >= 16);
        const int  m    = lane & 15;        // A-row / B-column within tile
        const int  n    = nb + m;           // fc2 output channel for this lane

        v8f acc = {};
        #pragma unroll 4
        for (int s = 0; s < 16; ++s) {
            const int kb = s * 32;
            // A fragment (16-bit A 16x32 layout): lanes 0-15 hold K 0..7|16..23,
            // lanes 16-31 hold K 8..15|24..31, M = lane&15.
            const __bf16* arow = &hbuf[m * HSTRIDE + kb + (hiL ? 8 : 0)];
            v8bf alo = *(const v8bf*)(arow);
            v8bf ahi = *(const v8bf*)(arow + 16);
            v16bf A;
            #pragma unroll
            for (int i = 0; i < 8; ++i) { A[i] = alo[i]; A[i + 8] = ahi[i]; }
            // B fragment (32x16): lanes 0-15 hold K 0..7|16..23 of column n,
            // lanes 16-31 hold K 8..15|24..31; fc2_w is row-major over K.
            const __bf16* bcol = &fc2w_bf[n * TWO_H + kb + (hiL ? 8 : 0)];
            v8bf blo = *(const v8bf*)(bcol);
            v8bf bhi = *(const v8bf*)(bcol + 16);
            v16bf Bm;
            #pragma unroll
            for (int i = 0; i < 8; ++i) { Bm[i] = blo[i]; Bm[i + 8] = bhi[i]; }
            acc = __builtin_amdgcn_wmma_f32_16x16x32_bf16(
                false, A, false, Bm, (short)0, acc, false, false);
        }
        // C/D layout: VGPR r -> M = r + (laneHi ? 8 : 0), N = lane&15.
        const float bias = fc2b[n];
        #pragma unroll
        for (int r = 0; r < 8; ++r) {
            const int mm = r + (hiL ? 8 : 0);
            const float v = fmaxf(acc[r] + bias, 0.0f);
            h2buf[mm * H2STRIDE + n] = (__bf16)v;
        }
    }
    __syncthreads();

    // ---------------- fc3 = relu(h2 @ fc3_w.T + b): one WMMA per N tile ----
    if (wave < 2) {
        const __bf16* fc3w_bf = (const __bf16*)fc3w_u;
        const int  nb  = wave * 16;
        const bool hiL = (lane >= 16);
        const int  m   = lane & 15;
        const int  n   = nb + m;

        const __bf16* arow = &h2buf[m * H2STRIDE + (hiL ? 8 : 0)];
        v8bf alo = *(const v8bf*)(arow);
        v8bf ahi = *(const v8bf*)(arow + 16);
        v16bf A;
        #pragma unroll
        for (int i = 0; i < 8; ++i) { A[i] = alo[i]; A[i + 8] = ahi[i]; }
        const __bf16* bcol = &fc3w_bf[n * MLPH + (hiL ? 8 : 0)];
        v8bf blo = *(const v8bf*)(bcol);
        v8bf bhi = *(const v8bf*)(bcol + 16);
        v16bf Bm;
        #pragma unroll
        for (int i = 0; i < 8; ++i) { Bm[i] = blo[i]; Bm[i + 8] = bhi[i]; }

        v8f acc = {};
        acc = __builtin_amdgcn_wmma_f32_16x16x32_bf16(
            false, A, false, Bm, (short)0, acc, false, false);

        const float bias = fc3b[n];
        #pragma unroll
        for (int r = 0; r < 8; ++r) {
            const int mm = r + (hiL ? 8 : 0);
            h3buf[mm * H3STRIDE + n] = fmaxf(acc[r] + bias, 0.0f);
        }
    }
    __syncthreads();

    // ---------------- fc4: out[m] = h3[m] . fc4_w + fc4_b ------------------
    if (wave == 0 && lane < BT) {
        float s = fc4b[0];
        #pragma unroll
        for (int k = 0; k < MLPH; ++k)
            s += h3buf[lane * H3STRIDE + k] * fc4w[k];
        out[rbase + lane] = s;
    }
}

extern "C" void kernel_launch(void* const* d_in, const int* in_sizes, int n_in,
                              void* d_out, int out_size, void* d_ws, size_t ws_size,
                              hipStream_t stream) {
    const int*   idx0 = (const int*)d_in[0];
    const int*   idx1 = (const int*)d_in[1];
    const float* w1   = (const float*)d_in[2];
    const float* fc2w = (const float*)d_in[3];
    const float* fc2b = (const float*)d_in[4];
    const float* fc3w = (const float*)d_in[5];
    const float* fc3b = (const float*)d_in[6];
    const float* fc4w = (const float*)d_in[7];
    const float* fc4b = (const float*)d_in[8];
    float*       out  = (float*)d_out;

    // Workspace: bf16 copies of fc2_w (32x512) and fc3_w (32x32).
    unsigned short* ws2 = (unsigned short*)d_ws;
    unsigned short* ws3 = ws2 + MLPH * TWO_H;

    const int B = in_sizes[0] / NACT;   // 65536

    nnue_cvt_weights<<<(MLPH * TWO_H + 255) / 256, 256, 0, stream>>>(fc2w, fc3w, ws2, ws3);
    nnue_fused<<<B / BT, THREADS, 0, stream>>>(idx0, idx1, w1, ws2, fc2b,
                                               ws3, fc3b, fc4w, fc4b, out);
}
